// GOATLayer_7705171329694
// MI455X (gfx1250) — compile-verified, matrix operand order checked by reading, not run on hardware
//
#include <hip/hip_runtime.h>
#include <hip/hip_bf16.h>
#include <stdint.h>

// ---------------------------------------------------------------------------
// GOAT layer for MI455X (gfx1250, wave32, WMMA).
// N=4096, C=512, H=8, DH=64, E=131072, G=8.
//
// Attention is algebraically collapsed: K/V have only G=8 distinct rows
// (kv = vproj[batch]), so softmax over m reduces to an 8-way mixture with
// weights cnt_g * exp(q . k_g / sqrt(DH)).
//
// GEMMs run as bf16 WMMA (v_wmma_f32_16x16x32_bf16) with fp32 accumulate.
// Weights are pre-converted AND pre-transposed (Wt[Nc][K]) once, so both A
// and B fragments are two contiguous b128 global loads per lane (L2/WGP$
// resident weights) -> the GEMM inner loop has NO LDS and NO barriers.
// All fragments for a K-slice are loaded into distinct registers before the
// WMMA sequence so waits are staggered instead of full-drain per WMMA.
// ---------------------------------------------------------------------------

#define NN   4096
#define CC   512
#define HH   8
#define DHD  64
#define EE   131072
#define GG   8
#define FFC  (4 * CC)   // 2048

typedef __attribute__((ext_vector_type(16))) __bf16 v16bf;
typedef __attribute__((ext_vector_type(8)))  float  v8f;

// ---------------------------------------------------------------- utilities
__device__ __forceinline__ unsigned short bf16_rne(float f) {
  uint32_t u = __float_as_uint(f);
  u += 0x7fffu + ((u >> 16) & 1u);
  return (unsigned short)(u >> 16);
}

__global__ void zero_f32(float* __restrict__ p, long long n) {
  long long i = (long long)blockIdx.x * blockDim.x + threadIdx.x;
  long long s = (long long)gridDim.x * blockDim.x;
  for (; i < n; i += s) p[i] = 0.0f;
}

// fp32 -> bf16 (RNE), 4 elements per thread (b128 in / b64 out)
__global__ void f32_to_bf16(const float* __restrict__ in,
                            unsigned short* __restrict__ out, long long n4) {
  long long i = (long long)blockIdx.x * blockDim.x + threadIdx.x;
  long long s = (long long)gridDim.x * blockDim.x;
  for (; i < n4; i += s) {
    float4 f = ((const float4*)in)[i];
    ushort4 o;
    o.x = bf16_rne(f.x); o.y = bf16_rne(f.y);
    o.z = bf16_rne(f.z); o.w = bf16_rne(f.w);
    ((ushort4*)out)[i] = o;
  }
}

// fp32 [K,Nc] -> bf16 transposed [Nc,K], 32x32 LDS tile, coalesced both sides
__global__ __launch_bounds__(256) void f32_to_bf16_T(
    const float* __restrict__ in, unsigned short* __restrict__ out,
    int K, int Nc) {
  __shared__ float tile[32][33];
  const int kb = blockIdx.y * 32;
  const int cb = blockIdx.x * 32;
  const int tx = threadIdx.x & 31;
  const int ty = threadIdx.x >> 5;  // 0..7
#pragma unroll
  for (int i = 0; i < 32; i += 8)
    tile[ty + i][tx] = in[(size_t)(kb + ty + i) * Nc + cb + tx];
  __syncthreads();
#pragma unroll
  for (int i = 0; i < 32; i += 8)
    out[(size_t)(cb + ty + i) * K + kb + tx] = bf16_rne(tile[tx][ty + i]);
}

// ------------------------------------------------------------ WMMA GEMM
// out[M,Nc] = A[M,K](bf16, row-major) @ W (given as Wt[Nc,K] bf16) + bias.
// Block = 256 threads = 8 waves; tile = 128x64; no LDS, no barriers.
#define MBLK 128
#define NBLK 64
#define KSTEP 32

__global__ __launch_bounds__(256) void wmma_gemm_bf16(
    const unsigned short* __restrict__ A, const unsigned short* __restrict__ Wt,
    const float* __restrict__ bias, float* __restrict__ out,
    int M, int K, int Nc, int relu) {
  const int tid  = threadIdx.x;
  const int wave = tid >> 5;
  const int lane = tid & 31;
  const int lrow = lane & 15;
  const int khalf = (lane >> 4) << 3;  // 0 for lanes 0-15, 8 for lanes 16-31

  const int m_blk = blockIdx.y * MBLK;
  const int n_blk = blockIdx.x * NBLK;

  // 16-bit fragment layout (A and B identical shape after W transpose):
  // lanes 0-15 hold K 0..7 / 16..23, lanes 16-31 hold K 8..15 / 24..31.
  const unsigned short* aptr =
      A + (size_t)(m_blk + (wave << 4) + lrow) * K + khalf;
  const unsigned short* bptr0 = Wt + (size_t)(n_blk + lrow) * K + khalf;

  v8f acc[4] = {};

  for (int k0 = 0; k0 < K; k0 += KSTEP) {
    __builtin_prefetch(aptr + k0 + 2 * KSTEP, 0, 1);  // global_prefetch_b8

    // ---- load phase: A fragment + ALL 4 B fragments into distinct regs
    uint4 a01[2];
    a01[0] = *(const uint4*)(aptr + k0);
    a01[1] = *(const uint4*)(aptr + k0 + 16);

    uint4 bfrag[4][2];
#pragma unroll
    for (int j = 0; j < 4; ++j) {
      const unsigned short* bp = bptr0 + (size_t)(j << 4) * K + k0;
      bfrag[j][0] = *(const uint4*)(bp);
      bfrag[j][1] = *(const uint4*)(bp + 16);
    }

    // ---- compute phase: 4 WMMAs with staggered load waits
    v16bf af = __builtin_bit_cast(v16bf, a01);
#pragma unroll
    for (int j = 0; j < 4; ++j) {
      v16bf bfr = __builtin_bit_cast(v16bf, bfrag[j]);
      acc[j] = __builtin_amdgcn_wmma_f32_16x16x32_bf16(
          false, af, false, bfr, (short)0, acc[j], false, false);
    }
  }

  // Epilogue. D layout: VGPR r -> row = r (+8 for lanes 16-31), col = lane%16.
  const int rbase = m_blk + (wave << 4) + khalf;
#pragma unroll
  for (int j = 0; j < 4; ++j) {
    int col  = n_blk + (j << 4) + lrow;
    float bv = bias[col];
#pragma unroll
    for (int r = 0; r < 8; ++r) {
      float v = acc[j][r] + bv;
      if (relu) v = fmaxf(v, 0.0f);
      out[(size_t)(rbase + r) * Nc + col] = v;
    }
  }
}

// -------------------------------------------------------- edge aggregation
__global__ void deg_kernel(const int* __restrict__ ei, float* __restrict__ deg) {
  int i = blockIdx.x * blockDim.x + threadIdx.x;
  int s = gridDim.x * blockDim.x;
  for (; i < EE; i += s) atomicAdd(&deg[ei[EE + i]], 1.0f);
}

// agg[dst] += msg[src] ; float4 per work-item, lanes cover consecutive channels
__global__ void edge_scatter(const int* __restrict__ ei,
                             const float* __restrict__ msg,
                             float* __restrict__ agg) {
  long long total = (long long)EE * (CC / 4);
  long long i = (long long)blockIdx.x * blockDim.x + threadIdx.x;
  long long st = (long long)gridDim.x * blockDim.x;
  for (; i < total; i += st) {
    int e  = (int)(i >> 7);
    int c4 = (int)(i & 127);
    int s = ei[e];
    int d = ei[EE + e];
    float4 m = ((const float4*)(msg + (size_t)s * CC))[c4];
    float* ap = agg + (size_t)d * CC + c4 * 4;
    atomicAdd(ap + 0, m.x);
    atomicAdd(ap + 1, m.y);
    atomicAdd(ap + 2, m.z);
    atomicAdd(ap + 3, m.w);
  }
}

// ---------------------------------------------------------------- layernorm
// out = LN(a + b * scale), scale = 1/(deg[row]+1e-8) if deg != null else 1.
__global__ __launch_bounds__(256) void ln_kernel(
    const float* __restrict__ a, const float* __restrict__ b,
    const float* __restrict__ deg, const float* __restrict__ gamma,
    const float* __restrict__ beta, float* __restrict__ out) {
  const int row = blockIdx.x;
  const int t   = threadIdx.x;  // 256 threads, C=512 -> 2 per thread
  float scale = 1.0f;
  if (deg) scale = 1.0f / (deg[row] + 1e-8f);
  size_t base = (size_t)row * CC;
  float v0 = a[base + t]       + b[base + t]       * scale;
  float v1 = a[base + t + 256] + b[base + t + 256] * scale;

  __shared__ float s_sum[256];
  __shared__ float s_sq[256];
  s_sum[t] = v0 + v1;
  s_sq[t]  = v0 * v0 + v1 * v1;
  __syncthreads();
  for (int off = 128; off > 0; off >>= 1) {
    if (t < off) { s_sum[t] += s_sum[t + off]; s_sq[t] += s_sq[t + off]; }
    __syncthreads();
  }
  float mu  = s_sum[0] * (1.0f / CC);
  float var = s_sq[0] * (1.0f / CC) - mu * mu;
  float inv = rsqrtf(var + 1e-5f);
  out[base + t]       = (v0 - mu) * inv * gamma[t]       + beta[t];
  out[base + t + 256] = (v1 - mu) * inv * gamma[t + 256] + beta[t + 256];
}

// ------------------------------------------------------------ graph pooling
__global__ void seg_cnt(const int* __restrict__ batch, float* __restrict__ vcnt) {
  int n = blockIdx.x * blockDim.x + threadIdx.x;
  if (n < NN) atomicAdd(&vcnt[batch[n]], 1.0f);
}

__global__ void seg_sum(const int* __restrict__ batch,
                        const float* __restrict__ x, float* __restrict__ vsum) {
  long long total = (long long)NN * (CC / 4);
  long long i = (long long)blockIdx.x * blockDim.x + threadIdx.x;
  long long st = (long long)gridDim.x * blockDim.x;
  for (; i < total; i += st) {
    int n  = (int)(i >> 7);
    int c4 = (int)(i & 127);
    int g  = batch[n];
    float4 v = ((const float4*)(x + (size_t)n * CC))[c4];
    float* vp = vsum + (size_t)g * CC + c4 * 4;
    atomicAdd(vp + 0, v.x);
    atomicAdd(vp + 1, v.y);
    atomicAdd(vp + 2, v.z);
    atomicAdd(vp + 3, v.w);
  }
}

// out[g,c] = (A[g,:]*scale_g) . W[:,c] + bias[c] ; scale_g = 1/max(cnt[g],1)
__global__ void graph_gemm(const float* __restrict__ A,
                           const float* __restrict__ cnt,
                           const float* __restrict__ W,
                           const float* __restrict__ bias,
                           float* __restrict__ out) {
  int t = blockIdx.x * blockDim.x + threadIdx.x;
  if (t >= GG * CC) return;
  int g = t / CC, c = t % CC;
  float scale = 1.0f;
  if (cnt) scale = 1.0f / fmaxf(cnt[g], 1.0f);
  float acc = bias[c];
  const float* ar = A + (size_t)g * CC;
  for (int k = 0; k < CC; ++k) acc += ar[k] * scale * W[(size_t)k * CC + c];
  out[t] = acc;
}

// -------------------------------------------------- collapsed attention
// ctx[n,h,:] = sum_g  w_g * v_g[h,:] / Z,  w_g = cnt_g * exp(q.k_g/8 - max)
__global__ void attn_kernel(const float* __restrict__ q,
                            const float* __restrict__ kg,
                            const float* __restrict__ vg,
                            const float* __restrict__ cnt,
                            float* __restrict__ ctx) {
  int t = blockIdx.x * blockDim.x + threadIdx.x;  // N*H threads
  if (t >= NN * HH) return;
  int n = t / HH, h = t % HH;
  const float* qp = q + (size_t)n * CC + h * DHD;

  float s[GG];
#pragma unroll
  for (int g = 0; g < GG; ++g) s[g] = 0.0f;
  for (int d = 0; d < DHD; ++d) {
    float qd = qp[d];
#pragma unroll
    for (int g = 0; g < GG; ++g) s[g] += qd * kg[(size_t)g * CC + h * DHD + d];
  }
  float m = -3.4e38f;
#pragma unroll
  for (int g = 0; g < GG; ++g) { s[g] *= 0.125f; m = fmaxf(m, s[g]); }
  float w[GG];
  float Z = 0.0f;
#pragma unroll
  for (int g = 0; g < GG; ++g) { w[g] = cnt[g] * __expf(s[g] - m); Z += w[g]; }
  float iz = 1.0f / Z;

  float* op = ctx + (size_t)n * CC + h * DHD;
  for (int d = 0; d < DHD; ++d) {
    float o = 0.0f;
#pragma unroll
    for (int g = 0; g < GG; ++g) o += w[g] * vg[(size_t)g * CC + h * DHD + d];
    op[d] = o * iz;
  }
}

// ---------------------------------------------------------------------------
extern "C" void kernel_launch(void* const* d_in, const int* in_sizes, int n_in,
                              void* d_out, int out_size, void* d_ws, size_t ws_size,
                              hipStream_t stream) {
  const float* x     = (const float*)d_in[0];
  const int*   ei    = (const int*)d_in[1];
  const int*   batch = (const int*)d_in[2];
  const float* w_msg = (const float*)d_in[4];
  const float* b_msg = (const float*)d_in[5];
  const float* wq    = (const float*)d_in[6];
  const float* bq    = (const float*)d_in[7];
  const float* wk    = (const float*)d_in[8];
  const float* bk    = (const float*)d_in[9];
  const float* wv    = (const float*)d_in[10];
  const float* bv    = (const float*)d_in[11];
  const float* wo    = (const float*)d_in[12];
  const float* bo    = (const float*)d_in[13];
  const float* w_vp  = (const float*)d_in[14];
  const float* b_vp  = (const float*)d_in[15];
  const float* g1 = (const float*)d_in[16]; const float* be1 = (const float*)d_in[17];
  const float* g2 = (const float*)d_in[18]; const float* be2 = (const float*)d_in[19];
  const float* g3 = (const float*)d_in[20]; const float* be3 = (const float*)d_in[21];
  const float* w1  = (const float*)d_in[22]; const float* bf1 = (const float*)d_in[23];
  const float* w2  = (const float*)d_in[24]; const float* bf2 = (const float*)d_in[25];

  // workspace carve-out (256B aligned)
  char* cur = (char*)d_ws;
  auto take = [&](size_t bytes) {
    char* p = cur;
    cur += (bytes + 255) & ~(size_t)255;
    return (void*)p;
  };
  unsigned short* bf_x   = (unsigned short*)take((size_t)NN * CC * 2);   // 4 MiB
  unsigned short* bf_w   = (unsigned short*)take((size_t)CC * FFC * 2);  // 2 MiB (transposed weights)
  unsigned short* bf_ffn = (unsigned short*)take((size_t)NN * FFC * 2);  // 16 MiB
  float* fA = (float*)take((size_t)NN * CC * 4);   // msg -> x_global -> ffn2
  float* fB = (float*)take((size_t)NN * CC * 4);   // agg -> ctx
  float* fC = (float*)take((size_t)NN * CC * 4);   // x1 (post-LN1 residual)
  float* fD = (float*)take((size_t)NN * CC * 4);   // q -> x2 (post-LN2 residual)
  float* fF = (float*)take((size_t)NN * FFC * 4);  // ffn hidden (32 MiB)
  float* deg   = (float*)take((size_t)NN * 4);
  float* vsum  = (float*)take((size_t)GG * CC * 4);
  float* vcnt  = (float*)take((size_t)GG * 4);
  float* vproj = (float*)take((size_t)GG * CC * 4);
  float* kgbuf = (float*)take((size_t)GG * CC * 4);
  float* vgbuf = (float*)take((size_t)GG * CC * 4);

  const dim3 blk(256);
  const dim3 cgrid(1024);                        // grid-stride converters (x4)
  const dim3 gemm_c(CC / NBLK, NN / MBLK);       // (8, 32)
  const dim3 gemm_f1(FFC / NBLK, NN / MBLK);     // (32, 32)
  const dim3 tr_cc(CC / 32, CC / 32);            // 512x512 transpose
  const dim3 tr_w1(FFC / 32, CC / 32);           // [512,2048] -> [2048,512]
  const dim3 tr_w2(CC / 32, FFC / 32);           // [2048,512] -> [512,2048]

  // 1) msg = x @ w_msg + b_msg   (bf16 WMMA, transposed weight)
  f32_to_bf16<<<cgrid, blk, 0, stream>>>(x, bf_x, (long long)NN * CC / 4);
  f32_to_bf16_T<<<tr_cc, blk, 0, stream>>>(w_msg, bf_w, CC, CC);
  wmma_gemm_bf16<<<gemm_c, blk, 0, stream>>>(bf_x, bf_w, b_msg, fA, NN, CC, CC, 0);

  // 2) scatter-add over edges + degree
  zero_f32<<<dim3(2048), blk, 0, stream>>>(fB, (long long)NN * CC);
  zero_f32<<<dim3(16), blk, 0, stream>>>(deg, NN);
  zero_f32<<<dim3(16), blk, 0, stream>>>(vsum, GG * CC);
  zero_f32<<<dim3(1), blk, 0, stream>>>(vcnt, GG);
  deg_kernel<<<dim3(512), blk, 0, stream>>>(ei, deg);
  edge_scatter<<<dim3(4096), blk, 0, stream>>>(ei, fA, fB);

  // 3) x1 = LN(x + agg/(deg+1e-8))
  ln_kernel<<<dim3(NN), blk, 0, stream>>>(x, fB, deg, g1, be1, fC);

  // 4) q = x1 @ wq + bq
  f32_to_bf16<<<cgrid, blk, 0, stream>>>(fC, bf_x, (long long)NN * CC / 4);
  f32_to_bf16_T<<<tr_cc, blk, 0, stream>>>(wq, bf_w, CC, CC);
  wmma_gemm_bf16<<<gemm_c, blk, 0, stream>>>(bf_x, bf_w, bq, fD, NN, CC, CC, 0);

  // 5) per-graph pooling; vproj = vmean @ w_vp + b_vp; k_g, v_g projections
  seg_cnt<<<dim3(16), blk, 0, stream>>>(batch, vcnt);
  seg_sum<<<dim3(2048), blk, 0, stream>>>(batch, fC, vsum);
  graph_gemm<<<dim3(16), blk, 0, stream>>>(vsum, vcnt, w_vp, b_vp, vproj);
  graph_gemm<<<dim3(16), blk, 0, stream>>>(vproj, nullptr, wk, bk, kgbuf);
  graph_gemm<<<dim3(16), blk, 0, stream>>>(vproj, nullptr, wv, bv, vgbuf);

  // 6) collapsed 8-way attention mixture -> ctx
  attn_kernel<<<dim3(NN * HH / 256), blk, 0, stream>>>(fD, kgbuf, vgbuf, vcnt, fB);

  // 7) x_global = ctx @ wo + bo ; x2 = LN(x1 + x_global)
  f32_to_bf16<<<cgrid, blk, 0, stream>>>(fB, bf_x, (long long)NN * CC / 4);
  f32_to_bf16_T<<<tr_cc, blk, 0, stream>>>(wo, bf_w, CC, CC);
  wmma_gemm_bf16<<<gemm_c, blk, 0, stream>>>(bf_x, bf_w, bo, fA, NN, CC, CC, 0);
  ln_kernel<<<dim3(NN), blk, 0, stream>>>(fC, fA, nullptr, g2, be2, fD);

  // 8) FFN: relu(x2 @ w1 + bf1) @ w2 + bf2
  f32_to_bf16<<<cgrid, blk, 0, stream>>>(fD, bf_x, (long long)NN * CC / 4);
  f32_to_bf16_T<<<tr_w1, blk, 0, stream>>>(w1, bf_w, CC, FFC);
  wmma_gemm_bf16<<<gemm_f1, blk, 0, stream>>>(bf_x, bf_w, bf1, fF, NN, CC, FFC, 1);
  f32_to_bf16<<<dim3(4096), blk, 0, stream>>>(fF, bf_ffn, (long long)NN * FFC / 4);
  f32_to_bf16_T<<<tr_w2, blk, 0, stream>>>(w2, bf_w, FFC, CC);
  wmma_gemm_bf16<<<gemm_c, blk, 0, stream>>>(bf_ffn, bf_w, bf2, fA, NN, FFC, CC, 0);

  // 9) out = LN(x2 + ffn)
  ln_kernel<<<dim3(NN), blk, 0, stream>>>(fD, fA, nullptr, g3, be3, (float*)d_out);

  (void)in_sizes; (void)n_in; (void)out_size; (void)ws_size;
}
